// SelfAttention_79869211836492
// MI455X (gfx1250) — matrix operand order repaired and checked by reading, not compiled, not run on hardware
//
#include <hip/hip_runtime.h>
#include <hip/hip_bf16.h>

#ifndef __has_builtin
#define __has_builtin(x) 0
#endif

#define B_   2
#define S_   2048
#define DM   1024
#define NH   16
#define DK   64
#define TOK  (B_ * S_)     // 4096 token rows
#define SCALE_ 0.125f      // 1/sqrt(64)
#define LDP  40            // padded LDS row stride in elements (80B = 5x16B)

typedef __bf16 bf16x16 __attribute__((ext_vector_type(16)));
typedef float  f32x8   __attribute__((ext_vector_type(8)));
typedef unsigned int   u32x4 __attribute__((ext_vector_type(4)));
typedef int            i32x4 __attribute__((ext_vector_type(4)));
typedef int            i32x8 __attribute__((ext_vector_type(8)));
typedef unsigned short u16;

union Frag { bf16x16 v; uint4 u[2]; };

__device__ __forceinline__ u16 f2bf(float x) {
    unsigned u = __float_as_uint(x);
    u += 0x7FFFu + ((u >> 16) & 1u);            // round-to-nearest-even
    return (u16)(u >> 16);
}
__device__ __forceinline__ float bf2f(u16 x) { return __uint_as_float(((unsigned)x) << 16); }

// ---- 16-lane all-reduce via DPP16 butterfly (no LDS, no dscnt waits) ----
// stages: xor1 (quad_perm 1,0,3,2), xor2 (quad_perm 2,3,0,1),
//         xor7 (row_half_mirror 0x141), xor15 (row_mirror 0x140)
#define DPPF(x, ctrl) \
    __int_as_float(__builtin_amdgcn_update_dpp(0, __float_as_int(x), (ctrl), 0xf, 0xf, true))

__device__ __forceinline__ float redmax16(float x) {
    x = fmaxf(x, DPPF(x, 0x0B1));
    x = fmaxf(x, DPPF(x, 0x04E));
    x = fmaxf(x, DPPF(x, 0x141));
    x = fmaxf(x, DPPF(x, 0x140));
    return x;
}
__device__ __forceinline__ float redsum16(float x) {
    x += DPPF(x, 0x0B1);
    x += DPPF(x, 0x04E);
    x += DPPF(x, 0x141);
    x += DPPF(x, 0x140);
    return x;
}

// A-fragment 16x32 (row-major [16][ld]); ISA 16-bit A layout:
// lanes 0-15: row=l, K {0..7,16..23}; lanes 16-31: row=l-16, K {8..15,24..31}
__device__ __forceinline__ Frag load_a(const u16* base, int ld) {
    const int l = threadIdx.x & 31;
    const u16* p = base + (size_t)(l & 15) * ld + ((l < 16) ? 0 : 8);
    Frag f;
    f.u[0] = *(const uint4*)(p);
    f.u[1] = *(const uint4*)(p + 16);
    return f;
}
// B-fragment 32x16, source stored as row-major [16 n][ld k] (i.e. B^T):
// lane holds fixed n=l&15; lanes 0-15 carry K {0..7,16..23}, lanes 16-31 carry K {8..15,24..31}
__device__ __forceinline__ Frag load_b(const u16* base, int ld) {
    const int l = threadIdx.x & 31;
    const u16* p = base + (size_t)(l & 15) * ld + ((l < 16) ? 0 : 8);
    Frag f;
    f.u[0] = *(const uint4*)(p);
    f.u[1] = *(const uint4*)(p + 16);
    return f;
}
__device__ __forceinline__ f32x8 wmma_bf16(Frag a, Frag b, f32x8 c) {
    return __builtin_amdgcn_wmma_f32_16x16x32_bf16(false, a.v, false, b.v, (short)0, c, false, false);
}

// ---- Tensor Data Mover: 2-D tile (bf16) -> LDS with row padding ----
// D# per CDNA5 ISA: group0 {count=1, lds_addr, global_addr, type=2},
// group1 {data_size=2B, pad_enable, pad_interval=16 DW (code 3), pad_amount=4 DW (code 3),
//         tensor_dim0/1, tile_dim0/1, tensor_dim0_stride}.  Pads +16B per 64B row
// => LDS row stride 40 elements, matching As/Bs/LDP exactly.
// This toolchain declares the 6-arg (clang-23 / therock) form:
//   (uint32x4 g0, int32x8 g1, int32x4 g2, int32x4 g3, int32x8 extra, i32 cpol)
#if __has_builtin(__builtin_amdgcn_tensor_load_to_lds)
#define HAVE_TDM 1
__device__ __forceinline__ void tdm_load_2d(unsigned lds_addr, const void* gaddr,
                                            unsigned tensor_d0, unsigned tensor_d1,
                                            unsigned tile_d0, unsigned tile_d1,
                                            unsigned stride_elems) {
    const unsigned long long ga = (unsigned long long)(uintptr_t)gaddr;
    u32x4 g0;
    g0[0] = 1u;                                                   // count=1 (valid), user mode
    g0[1] = lds_addr;                                             // LDS byte address
    g0[2] = (unsigned)(ga & 0xFFFFFFFFu);                         // global_addr[31:0]
    g0[3] = (unsigned)((ga >> 32) & 0x01FFFFFFu) | (2u << 30);    // global_addr[56:32], type=2
    i32x8 g1;
    g1[0] = (int)((1u << 16) | (1u << 20) | (3u << 22) | (3u << 25)); // 2B, pad 4DW/16DW
    g1[1] = (int)((tensor_d0 & 0xFFFFu) << 16);                   // tensor_dim0[15:0]
    g1[2] = (int)((tensor_d0 >> 16) | ((tensor_d1 & 0xFFFFu) << 16));
    g1[3] = (int)((tensor_d1 >> 16) | (tile_d0 << 16));           // tile_dim0
    g1[4] = (int)(tile_d1 & 0xFFFFu);                             // tile_dim1 (tile_dim2=0)
    g1[5] = (int)stride_elems;                                    // tensor_dim0_stride[31:0]
    g1[6] = 0;
    g1[7] = 0;
    i32x4 z4 = {0, 0, 0, 0};                                      // groups 2/3 unused (<=2-D)
    i32x8 z8 = {0, 0, 0, 0, 0, 0, 0, 0};
    __builtin_amdgcn_tensor_load_to_lds(g0, g1, z4, z4, z8, 0);
}
#else
#define HAVE_TDM 0
#endif

// ---------------- small prep kernels ----------------

__global__ void k_bias_concat(const float* __restrict__ bq, const float* __restrict__ bk,
                              const float* __restrict__ bv, float* __restrict__ out) {
    int i = blockIdx.x * blockDim.x + threadIdx.x;
    if (i < DM)            out[i] = bq[i];
    else if (i < 2 * DM)   out[i] = bk[i - DM];
    else if (i < 3 * DM)   out[i] = bv[i - 2 * DM];
}

// dst[n][k] = (bf16) src[k][n]   (1024x1024)
__global__ void k_transpose_bf16(const float* __restrict__ src, u16* __restrict__ dst) {
    __shared__ float tile[32][33];
    const int tx = threadIdx.x, ty = threadIdx.y;
    const int x  = blockIdx.x * 32 + tx;
    const int y0 = blockIdx.y * 32;
    for (int j = 0; j < 32; j += 8)
        tile[ty + j][tx] = src[(size_t)(y0 + ty + j) * DM + x];
    __syncthreads();
    const int ox = blockIdx.y * 32 + tx;
    const int oy = blockIdx.x * 32;
    for (int j = 0; j < 32; j += 8)
        dst[(size_t)(oy + ty + j) * DM + ox] = f2bf(tile[tx][ty + j]);
}

__global__ void k_rmsnorm(const float* __restrict__ h, const float* __restrict__ g,
                          u16* __restrict__ hn) {
    const int row = blockIdx.x;
    const int t   = threadIdx.x;
    const float4 hv = *(const float4*)(h + (size_t)row * DM + t * 4);
    float ss = hv.x * hv.x + hv.y * hv.y + hv.z * hv.z + hv.w * hv.w;
    ss = redsum16(ss);
    ss += __shfl_xor(ss, 16);                 // one cross-half step
    __shared__ float red[8];
    if ((t & 31) == 0) red[t >> 5] = ss;
    __syncthreads();
    float tot = 0.f;
#pragma unroll
    for (int i = 0; i < 8; i++) tot += red[i];
    const float rn = rsqrtf(tot * (1.0f / DM) + 1e-6f);
    const float4 gv = *(const float4*)(g + t * 4);
    ushort4 ov = make_ushort4(f2bf(hv.x * rn * gv.x), f2bf(hv.y * rn * gv.y),
                              f2bf(hv.z * rn * gv.z), f2bf(hv.w * rn * gv.w));
    *(ushort4*)(hn + (size_t)row * DM + t * 4) = ov;
}

// ---------------- tiled WMMA GEMM: C[M][N] = A[M][K] * Bt[N][K]^T + bias (+res) ----------------
// block tile 64x128, 8 waves in 2x4, each wave 32x32 (2x2 WMMA tiles), K-step 32.
// Tiles streamed into LDS by the Tensor Data Mover, double buffered; wave 0 issues,
// s_wait_tensorcnt + workgroup barrier orders consumers.

template <bool OUTBF16>
__global__ void k_gemm(const u16* __restrict__ A, const u16* __restrict__ Bt,
                       const float* __restrict__ bias, const float* __restrict__ residual,
                       void* __restrict__ out, int M, int N, int K) {
    __shared__ __align__(16) u16 As[2][64][LDP];
    __shared__ __align__(16) u16 Bs[2][128][LDP];
    const int t  = threadIdx.x;
    const int w  = t >> 5;
    const int wm = w >> 2, wn = w & 3;
    const int rowbase = blockIdx.y * 64;
    const int colbase = blockIdx.x * 128;
    const int KT = K >> 5;

    f32x8 c00 = {}, c01 = {}, c10 = {}, c11 = {};

#if HAVE_TDM
    if (t < 32) {   // wave 0 drives the TDM
        tdm_load_2d((unsigned)(uintptr_t)&As[0][0][0], A + (size_t)rowbase * K,
                    (unsigned)K, (unsigned)M, 32u, 64u, (unsigned)K);
        tdm_load_2d((unsigned)(uintptr_t)&Bs[0][0][0], Bt + (size_t)colbase * K,
                    (unsigned)K, (unsigned)N, 32u, 128u, (unsigned)K);
    }
    for (int kt = 0; kt < KT; ++kt) {
        const int buf = kt & 1;
        if (t < 32) {
            if (kt + 1 < KT) {
                const int k1 = (kt + 1) * 32;
                tdm_load_2d((unsigned)(uintptr_t)&As[buf ^ 1][0][0],
                            A + (size_t)rowbase * K + k1,
                            (unsigned)K, (unsigned)M, 32u, 64u, (unsigned)K);
                tdm_load_2d((unsigned)(uintptr_t)&Bs[buf ^ 1][0][0],
                            Bt + (size_t)colbase * K + k1,
                            (unsigned)K, (unsigned)N, 32u, 128u, (unsigned)K);
                __builtin_amdgcn_s_wait_tensorcnt(2);   // current pair landed (in-order)
            } else {
                __builtin_amdgcn_s_wait_tensorcnt(0);
            }
        }
        __syncthreads();
        Frag a0 = load_a(&As[buf][wm * 32][0], LDP);
        Frag a1 = load_a(&As[buf][wm * 32 + 16][0], LDP);
        Frag b0 = load_b(&Bs[buf][wn * 32][0], LDP);
        Frag b1 = load_b(&Bs[buf][wn * 32 + 16][0], LDP);
        c00 = wmma_bf16(a0, b0, c00);
        c01 = wmma_bf16(a0, b1, c01);
        c10 = wmma_bf16(a1, b0, c10);
        c11 = wmma_bf16(a1, b1, c11);
        __syncthreads();   // buffer may be overwritten next iteration
    }
#else
    const int ar = t >> 2, ac = (t & 3) * 8;
    const int br = t >> 1, bc = (t & 1) * 16;
    for (int kt = 0; kt < KT; ++kt) {
        const int k0 = kt * 32;
        *(uint4*)&As[0][ar][ac]     = *(const uint4*)(A  + (size_t)(rowbase + ar) * K + k0 + ac);
        *(uint4*)&Bs[0][br][bc]     = *(const uint4*)(Bt + (size_t)(colbase + br) * K + k0 + bc);
        *(uint4*)&Bs[0][br][bc + 8] = *(const uint4*)(Bt + (size_t)(colbase + br) * K + k0 + bc + 8);
        if (kt + 1 < KT) {
            __builtin_prefetch(A  + (size_t)(rowbase + ar) * K + k0 + 32 + ac, 0, 0);
            __builtin_prefetch(Bt + (size_t)(colbase + br) * K + k0 + 32 + bc, 0, 0);
        }
        __syncthreads();
        Frag a0 = load_a(&As[0][wm * 32][0], LDP);
        Frag a1 = load_a(&As[0][wm * 32 + 16][0], LDP);
        Frag b0 = load_b(&Bs[0][wn * 32][0], LDP);
        Frag b1 = load_b(&Bs[0][wn * 32 + 16][0], LDP);
        c00 = wmma_bf16(a0, b0, c00);
        c01 = wmma_bf16(a0, b1, c01);
        c10 = wmma_bf16(a1, b0, c10);
        c11 = wmma_bf16(a1, b1, c11);
        __syncthreads();
    }
#endif

    const int l    = t & 31;
    const int nl   = l & 15;
    const int roff = (l < 16) ? 0 : 8;
    f32x8 cc[2][2] = {{c00, c01}, {c10, c11}};
    for (int mi = 0; mi < 2; ++mi)
        for (int ni = 0; ni < 2; ++ni)
#pragma unroll
            for (int r = 0; r < 8; ++r) {
                const int row = rowbase + wm * 32 + mi * 16 + r + roff;
                const int col = colbase + wn * 32 + ni * 16 + nl;
                float v = cc[mi][ni][r] + bias[col];
                if (residual) v += residual[(size_t)row * N + col];
                if (OUTBF16) ((u16*)out)[(size_t)row * N + col] = f2bf(v);
                else         ((float*)out)[(size_t)row * N + col] = v;
            }
}

// ---------------- RoPE + attention-friendly layouts ----------------
// qkv: bf16 [4096][3072] (bias already added).  Writes:
//   Qh: [B][H][S][64] roped AND pre-scaled by 1/sqrt(dk)
//   Kh: [B][H][S][64] roped,  Vt: [B][H][64][S] transposed for B-fragments

__global__ void k_rope(const u16* __restrict__ qkv, u16* __restrict__ Qh,
                       u16* __restrict__ Kh, u16* __restrict__ Vt) {
    const int idx = blockIdx.x * blockDim.x + threadIdx.x;   // TOK*NH*32
    const int j  = idx & 31;
    const int hh = (idx >> 5) & (NH - 1);
    const int tk = idx >> 9;
    const int b  = tk / S_, s = tk - b * S_;
    const u16* row = qkv + (size_t)tk * (3 * DM);
    const float q0 = bf2f(row[hh * DK + j]),          q1 = bf2f(row[hh * DK + j + 32]);
    const float k0 = bf2f(row[DM + hh * DK + j]),     k1 = bf2f(row[DM + hh * DK + j + 32]);
    const float v0 = bf2f(row[2 * DM + hh * DK + j]), v1 = bf2f(row[2 * DM + hh * DK + j + 32]);
    // theta_j = BASE^(-2j/64) ; ln(10000)=9.2103403719...
    const float ang = (float)s * __expf(-9.210340371976184f * (float)(2 * j) * (1.0f / 64.0f));
    float sn, cs;
    __sincosf(ang, &sn, &cs);
    const size_t qb = ((size_t)(b * NH + hh) * S_ + s) * DK;
    Qh[qb + j]      = f2bf((q0 * cs - q1 * sn) * SCALE_);
    Qh[qb + j + 32] = f2bf((q1 * cs + q0 * sn) * SCALE_);
    Kh[qb + j]      = f2bf(k0 * cs - k1 * sn);
    Kh[qb + j + 32] = f2bf(k1 * cs + k0 * sn);
    const size_t vb = (size_t)(b * NH + hh) * DK * S_;
    Vt[vb + (size_t)j * S_ + s]        = f2bf(v0);
    Vt[vb + (size_t)(j + 32) * S_ + s] = f2bf(v1);
}

// ---------------- flash attention: 1 wave = 16 query rows, key tiles of 32 ----------------

__global__ void k_attention(const u16* __restrict__ Qh, const u16* __restrict__ Kh,
                            const u16* __restrict__ Vt, u16* __restrict__ attn) {
    __shared__ __align__(16) u16 pbuf[8][16][LDP];   // per-wave P staging (C-layout -> A-layout)
    const int w    = threadIdx.x >> 5;
    const int l    = threadIdx.x & 31;
    const int nl   = l & 15;
    const int roff = (l < 16) ? 0 : 8;
    const int qblk = blockIdx.x & 15;                // S_/128 = 16 query blocks
    const int bh   = blockIdx.x >> 4;                // b*NH + h
    const int b    = bh >> 4, hh = bh & 15;
    const int qb   = qblk * 128 + w * 16;            // this wave's first query row

    const u16* Qp = Qh + (size_t)bh * S_ * DK;
    const u16* Kp = Kh + (size_t)bh * S_ * DK;
    const u16* Vp = Vt + (size_t)bh * DK * S_;

    const Frag qa0 = load_a(Qp + (size_t)qb * DK + 0, DK);
    const Frag qa1 = load_a(Qp + (size_t)qb * DK + 32, DK);

    f32x8 o0 = {}, o1 = {}, o2 = {}, o3 = {};
    float m[8], lsum[8];
#pragma unroll
    for (int r = 0; r < 8; ++r) { m[r] = -__builtin_inff(); lsum[r] = 0.f; }

    const int qmax = qb + 15;
    for (int jb = 0; jb <= qmax; jb += 32) {
        // scores: two 16x16 tiles (keys jb..+15, jb+16..+31), K-dim = d = 64
        f32x8 s0 = {}, s1 = {};
        s0 = wmma_bf16(qa0, load_b(Kp + (size_t)jb * DK + 0, DK), s0);
        s0 = wmma_bf16(qa1, load_b(Kp + (size_t)jb * DK + 32, DK), s0);
        s1 = wmma_bf16(qa0, load_b(Kp + (size_t)(jb + 16) * DK + 0, DK), s1);
        s1 = wmma_bf16(qa1, load_b(Kp + (size_t)(jb + 16) * DK + 32, DK), s1);

#pragma unroll
        for (int r = 0; r < 8; ++r) {
            const int i = qb + r + roff;
            float v0 = s0[r], v1 = s1[r];                    // Q was pre-scaled
            if (jb + nl > i)      v0 = -__builtin_inff();
            if (jb + 16 + nl > i) v1 = -__builtin_inff();
            const float mx   = redmax16(fmaxf(v0, v1));      // DPP butterfly, no LDS
            const float mnew = fmaxf(m[r], mx);
            const float corr = __expf(m[r] - mnew);
            const float p0 = __expf(v0 - mnew), p1 = __expf(v1 - mnew);
            const float rs = redsum16(p0 + p1);
            lsum[r] = lsum[r] * corr + rs;
            m[r] = mnew;
            o0[r] *= corr; o1[r] *= corr; o2[r] *= corr; o3[r] *= corr;
            pbuf[w][r + roff][nl]      = f2bf(p0);
            pbuf[w][r + roff][nl + 16] = f2bf(p1);
        }
        // P (16x32) as A-fragment from LDS, V B-fragments from Vt[d][s]
        const Frag pa = load_a(&pbuf[w][0][0], LDP);
        o0 = wmma_bf16(pa, load_b(Vp + (size_t)0  * S_ + jb, S_), o0);
        o1 = wmma_bf16(pa, load_b(Vp + (size_t)16 * S_ + jb, S_), o1);
        o2 = wmma_bf16(pa, load_b(Vp + (size_t)32 * S_ + jb, S_), o2);
        o3 = wmma_bf16(pa, load_b(Vp + (size_t)48 * S_ + jb, S_), o3);
    }
#pragma unroll
    for (int r = 0; r < 8; ++r) {
        const int i = qb + r + roff;
        const float inv = 1.0f / lsum[r];
        const size_t base = ((size_t)b * S_ + i) * DM + hh * DK;
        attn[base + 0 * 16 + nl] = f2bf(o0[r] * inv);
        attn[base + 1 * 16 + nl] = f2bf(o1[r] * inv);
        attn[base + 2 * 16 + nl] = f2bf(o2[r] * inv);
        attn[base + 3 * 16 + nl] = f2bf(o3[r] * inv);
    }
}

// ---------------- launch ----------------

extern "C" void kernel_launch(void* const* d_in, const int* in_sizes, int n_in,
                              void* d_out, int out_size, void* d_ws, size_t ws_size,
                              hipStream_t stream) {
    (void)in_sizes; (void)n_in; (void)out_size; (void)ws_size;
    const float* h  = (const float*)d_in[0];
    const float* wq = (const float*)d_in[1];
    const float* bq = (const float*)d_in[2];
    const float* wk = (const float*)d_in[3];
    const float* bk = (const float*)d_in[4];
    const float* wv = (const float*)d_in[5];
    const float* bv = (const float*)d_in[6];
    const float* wo = (const float*)d_in[7];
    const float* bo = (const float*)d_in[8];
    const float* g  = (const float*)d_in[9];

    char* p = (char*)d_ws;
    auto carve = [&](size_t bytes) -> void* {
        void* r = (void*)p;
        p += (bytes + 255) & ~(size_t)255;
        return r;
    };
    u16*   hn    = (u16*)carve((size_t)TOK * DM * 2);          // 8 MB
    u16*   wqkvT = (u16*)carve((size_t)3 * DM * DM * 2);       // 6 MB
    u16*   woT   = (u16*)carve((size_t)DM * DM * 2);           // 2 MB
    float* bqkv  = (float*)carve((size_t)3 * DM * 4);
    u16*   qkv   = (u16*)carve((size_t)TOK * 3 * DM * 2);      // 24 MB
    u16*   Qh    = (u16*)carve((size_t)TOK * DM * 2);          // 8 MB
    u16*   Kh    = (u16*)carve((size_t)TOK * DM * 2);          // 8 MB
    u16*   Vt    = (u16*)carve((size_t)TOK * DM * 2);          // 8 MB
    u16*   attnb = (u16*)carve((size_t)TOK * DM * 2);          // 8 MB

    k_bias_concat<<<dim3(12), dim3(256), 0, stream>>>(bq, bk, bv, bqkv);
    dim3 tg(32, 32), tb(32, 8);
    k_transpose_bf16<<<tg, tb, 0, stream>>>(wq, wqkvT);
    k_transpose_bf16<<<tg, tb, 0, stream>>>(wk, wqkvT + (size_t)DM * DM);
    k_transpose_bf16<<<tg, tb, 0, stream>>>(wv, wqkvT + (size_t)2 * DM * DM);
    k_transpose_bf16<<<tg, tb, 0, stream>>>(wo, woT);
    k_rmsnorm<<<TOK, 256, 0, stream>>>(h, g, hn);
    k_gemm<true><<<dim3(24, 64), 256, 0, stream>>>(hn, wqkvT, bqkv, nullptr, qkv,
                                                   TOK, 3 * DM, DM);
    k_rope<<<(TOK * NH * 32) / 256, 256, 0, stream>>>(qkv, Qh, Kh, Vt);
    k_attention<<<B_ * NH * (S_ / 128), 256, 0, stream>>>(Qh, Kh, Vt, attnb);
    k_gemm<false><<<dim3(8, 64), 256, 0, stream>>>(attnb, woT, bo, h, d_out,
                                                   TOK, DM, DM);
}
